// GaussianVectorQuantizer_18116172054716
// MI455X (gfx1250) — compile-verified, matrix operand order checked by reading, not run on hardware
//
#include <hip/hip_runtime.h>
#include <hip/hip_bf16.h>

typedef __attribute__((ext_vector_type(16))) __bf16 v16bf;
typedef __attribute__((ext_vector_type(8)))  float  v8f;
typedef __attribute__((ext_vector_type(4)))  unsigned int u32x4;
typedef unsigned short ushort_t;

union Frag { u32x4 u[2]; v16bf v; };

#define N_ROWS 4096
#define DIM    512
#define KCB    8192
#define BS     8
#define GHALF  (((unsigned)N_ROWS * (unsigned)KCB) >> 1)   /* 2^24 */

__device__ __forceinline__ ushort_t f2bf(float f) {
  unsigned u = __float_as_uint(f);
  unsigned r = u + 0x7fffu + ((u >> 16) & 1u);
  return (ushort_t)(r >> 16);
}

__device__ __forceinline__ float bf2f(ushort_t v) {
  return __uint_as_float((unsigned)v << 16);
}

__device__ __forceinline__ unsigned rotl32(unsigned x, int r) {
  return (x << r) | (x >> (32 - r));
}

// threefry2x32 core with key = JAX PRNGKey(42) -> (0, 42)
__device__ __forceinline__ void threefry_42(unsigned c0, unsigned c1,
                                            unsigned& o0, unsigned& o1) {
  const unsigned ks0 = 0u, ks1 = 42u, ks2 = 0x1BD11BDAu ^ ks0 ^ ks1;
  unsigned x0 = c0 + ks0, x1 = c1 + ks1;
#define TF4(ra,rb,rc,rd) \
  x0 += x1; x1 = rotl32(x1, ra); x1 ^= x0; \
  x0 += x1; x1 = rotl32(x1, rb); x1 ^= x0; \
  x0 += x1; x1 = rotl32(x1, rc); x1 ^= x0; \
  x0 += x1; x1 = rotl32(x1, rd); x1 ^= x0;
  TF4(13,15,26,6)  x0 += ks1; x1 += ks2 + 1u;
  TF4(17,29,16,24) x0 += ks2; x1 += ks0 + 2u;
  TF4(13,15,26,6)  x0 += ks0; x1 += ks1 + 3u;
  TF4(17,29,16,24) x0 += ks1; x1 += ks2 + 4u;
  TF4(13,15,26,6)  x0 += ks2; x1 += ks0 + 5u;
#undef TF4
  o0 = x0; o1 = x1;
}

__device__ __forceinline__ float gumbel_from_bits(unsigned bits) {
  float U = __uint_as_float((bits >> 9) | 0x3f800000u) - 1.0f;
  return -__logf(-__logf(U + 1e-10f) + 1e-10f);
}

// inline fallback (only used when workspace lacks room for the table):
// emitted once (noinline) so the unrolled callers don't clone ~70 VALU ops.
__device__ __noinline__ float gumbel_for(unsigned idx, bool hi) {
  unsigned base = hi ? (idx - GHALF) : idx;
  unsigned o0, o1;
  threefry_42(base, base + GHALF, o0, o1);
  return gumbel_from_bits(hi ? o1 : o0);
}

// ---------------- gumbel table precompute: 2 outputs per threefry block ----------------
__global__ __launch_bounds__(256)
void gumbel_tab_kernel(ushort_t* __restrict__ gtab) {
  unsigned i = blockIdx.x * 256u + threadIdx.x;   // i < GHALF
  unsigned o0, o1;
  threefry_42(i, i + GHALF, o0, o1);
  gtab[i]         = f2bf(gumbel_from_bits(o0));
  gtab[i + GHALF] = f2bf(gumbel_from_bits(o1));
}

// ---------------- prep kernels: f32 -> bf16 + row squared norms ----------------
__global__ __launch_bounds__(256)
void prep_z_kernel(const float* __restrict__ z, ushort_t* __restrict__ zf,
                   float* __restrict__ znorm) {
  __shared__ float red[256];
  const int row = blockIdx.x, tid = threadIdx.x;
  const float* src = z + (size_t)row * DIM;
  float s = 0.f;
#pragma unroll
  for (int r = 0; r < 2; ++r) {
    int d = tid + r * 256;
    float v = src[d];
    zf[(size_t)row * DIM + d] = f2bf(v);
    s += v * v;
  }
  red[tid] = s; __syncthreads();
  for (int off = 128; off > 0; off >>= 1) {
    if (tid < off) red[tid] += red[tid + off];
    __syncthreads();
  }
  if (tid == 0) znorm[row] = red[0];
}

__global__ __launch_bounds__(256)
void prep_cb_kernel(const float* __restrict__ c, ushort_t* __restrict__ cb,
                    ushort_t* __restrict__ cbT, float* __restrict__ cnorm) {
  __shared__ float red[256];
  const int row = blockIdx.x, tid = threadIdx.x;
  const float* src = c + (size_t)row * DIM;
  float s = 0.f;
#pragma unroll
  for (int r = 0; r < 2; ++r) {
    int d = tid + r * 256;
    float v = src[d];
    ushort_t b = f2bf(v);
    cb[(size_t)row * DIM + d] = b;
    cbT[(size_t)d * KCB + row] = b;   // k-major copy for the 2nd GEMM's B operand
    s += v * v;
  }
  red[tid] = s; __syncthreads();
  for (int off = 128; off > 0; off >>= 1) {
    if (tid < off) red[tid] += red[tid + off];
    __syncthreads();
  }
  if (tid == 0) cnorm[row] = red[0];
}

// ---------------- main fused VQ kernel: 16 rows per workgroup ----------------
__global__ __launch_bounds__(256)
void vq_main_kernel(const float* __restrict__ var_q,
                    const ushort_t* __restrict__ zf,
                    const ushort_t* __restrict__ cb,
                    const ushort_t* __restrict__ cbT,
                    const float* __restrict__ znorm,
                    const float* __restrict__ cnorm,
                    const ushort_t* __restrict__ gtab,   // may be null
                    float* __restrict__ avgp,
                    float* __restrict__ scal,
                    float* __restrict__ out) {
  __shared__ __align__(16) ushort_t E_lds[16 * 128];
  __shared__ float sm1[8][16], ss1[8][16], sm2[8][16], ss2[8][16];
  __shared__ float lse1_s[16], lse2_s[16];

  const int tid  = threadIdx.x;
  const int wid  = tid >> 5;
  const int lane = tid & 31;
  const int ml   = lane & 15;   // row (A) or col (B/D) within 16
  const int h    = lane >> 4;   // lane half
  const int row0 = blockIdx.x * 16;
  const bool hi_half = (row0 >= 2048);  // uniform: which threefry counter half

  const float w  = 0.5f / fmaxf(var_q[0], 1e-10f);
  const float w2 = 2.0f * w;

  float zn_r[8];
#pragma unroll
  for (int j = 0; j < 8; ++j) zn_r[j] = znorm[row0 + j + 8 * h];

  // preload the entire A row-tile fragment set (reused 128x across both phases)
  const ushort_t* zrow = zf + (size_t)(row0 + ml) * DIM;
  Frag afr[16];
#pragma unroll
  for (int dk = 0; dk < DIM; dk += 32) {
    afr[dk / 32].u[0] = *(const u32x4*)(zrow + dk + h * 8);
    afr[dk / 32].u[1] = *(const u32x4*)(zrow + dk + 16 + h * 8);
  }

  // ---------- Phase 1: per-lane online softmax stats ----------
  float rm1[8], rs1[8], rm2[8], rs2[8];
#pragma unroll
  for (int j = 0; j < 8; ++j) {
    rm1[j] = -__builtin_inff(); rs1[j] = 0.f;
    rm2[j] = -__builtin_inff(); rs2[j] = 0.f;
  }

  for (int kc = 0; kc < KCB; kc += 128) {
    const int kcol = kc + wid * 16 + ml;
    const ushort_t* crow = cb + (size_t)kcol * DIM;
    v8f acc = {0.f,0.f,0.f,0.f,0.f,0.f,0.f,0.f};
#pragma unroll
    for (int dk = 0; dk < DIM; dk += 32) {
      Frag b;
      b.u[0] = *(const u32x4*)(crow + dk + h * 8);
      b.u[1] = *(const u32x4*)(crow + dk + 16 + h * 8);
      acc = __builtin_amdgcn_wmma_f32_16x16x32_bf16(
          false, afr[dk / 32].v, false, b.v, (short)0, acc, false, false);
    }
    const float cn = cnorm[kcol];
#pragma unroll
    for (int j = 0; j < 8; ++j) {
      float l = w2 * acc[j] - w * (zn_r[j] + cn);      // -w * dist
      unsigned gidx = (unsigned)(row0 + j + 8 * h) * (unsigned)KCB + (unsigned)kcol;
      float g = gtab ? bf2f(gtab[gidx]) : gumbel_for(gidx, hi_half);
      float l2 = (l + g) * 2.0f;                        // / TEMPERATURE (0.5)
      float mo = rm1[j], mn = fmaxf(mo, l);
      rs1[j] = rs1[j] * __expf(mo - mn) + __expf(l - mn);  rm1[j] = mn;
      mo = rm2[j]; mn = fmaxf(mo, l2);
      rs2[j] = rs2[j] * __expf(mo - mn) + __expf(l2 - mn); rm2[j] = mn;
    }
  }

  // merge stats across the 16 lanes of each half
#pragma unroll
  for (int j = 0; j < 8; ++j) {
#pragma unroll
    for (int mask = 1; mask <= 8; mask <<= 1) {
      float mo = __shfl_xor(rm1[j], mask);
      float so = __shfl_xor(rs1[j], mask);
      float mn = fmaxf(rm1[j], mo);
      rs1[j] = rs1[j] * __expf(rm1[j] - mn) + so * __expf(mo - mn);
      rm1[j] = mn;
      mo = __shfl_xor(rm2[j], mask);
      so = __shfl_xor(rs2[j], mask);
      mn = fmaxf(rm2[j], mo);
      rs2[j] = rs2[j] * __expf(rm2[j] - mn) + so * __expf(mo - mn);
      rm2[j] = mn;
    }
  }
  if (ml == 0) {
#pragma unroll
    for (int j = 0; j < 8; ++j) {
      sm1[wid][j + 8 * h] = rm1[j]; ss1[wid][j + 8 * h] = rs1[j];
      sm2[wid][j + 8 * h] = rm2[j]; ss2[wid][j + 8 * h] = rs2[j];
    }
  }
  __syncthreads();
  if (tid < 16) {
    float m = -__builtin_inff(), s = 0.f;
    for (int wv = 0; wv < 8; ++wv) {
      float mo = sm1[wv][tid], so = ss1[wv][tid];
      float mn = fmaxf(m, mo);
      s = s * __expf(m - mn) + so * __expf(mo - mn); m = mn;
    }
    lse1_s[tid] = m + __logf(s);
    m = -__builtin_inff(); s = 0.f;
    for (int wv = 0; wv < 8; ++wv) {
      float mo = sm2[wv][tid], so = ss2[wv][tid];
      float mn = fmaxf(m, mo);
      s = s * __expf(m - mn) + so * __expf(mo - mn); m = mn;
    }
    lse2_s[tid] = m + __logf(s);
  }
  __syncthreads();
  float lse1_r[8], lse2_r[8];
#pragma unroll
  for (int j = 0; j < 8; ++j) {
    lse1_r[j] = lse1_s[j + 8 * h];
    lse2_r[j] = lse2_s[j + 8 * h];
  }

  // ---------- Phase 2: recompute logits, emit probs/encodings, 2nd GEMM ----------
  const v8f zerov = {0.f,0.f,0.f,0.f,0.f,0.f,0.f,0.f};
  v8f acc2[4] = {zerov, zerov, zerov, zerov};
  float kldd[8] = {0.f,0.f,0.f,0.f,0.f,0.f,0.f,0.f};

  for (int kc = 0; kc < KCB; kc += 128) {
    const int kcol = kc + wid * 16 + ml;
    const ushort_t* crow = cb + (size_t)kcol * DIM;
    v8f acc = {0.f,0.f,0.f,0.f,0.f,0.f,0.f,0.f};
#pragma unroll
    for (int dk = 0; dk < DIM; dk += 32) {
      Frag b;
      b.u[0] = *(const u32x4*)(crow + dk + h * 8);
      b.u[1] = *(const u32x4*)(crow + dk + 16 + h * 8);
      acc = __builtin_amdgcn_wmma_f32_16x16x32_bf16(
          false, afr[dk / 32].v, false, b.v, (short)0, acc, false, false);
    }
    const float cn = cnorm[kcol];
    float ps = 0.f;
#pragma unroll
    for (int j = 0; j < 8; ++j) {
      float l = w2 * acc[j] - w * (zn_r[j] + cn);
      unsigned gidx = (unsigned)(row0 + j + 8 * h) * (unsigned)KCB + (unsigned)kcol;
      float g = gtab ? bf2f(gtab[gidx]) : gumbel_for(gidx, hi_half);
      float l2 = (l + g) * 2.0f;
      float lp = l - lse1_r[j];
      float p  = __expf(lp);
      kldd[j] += p * lp;
      ps += p;
      float e = __expf(l2 - lse2_r[j]);            // gumbel-softmax weight
      E_lds[(j + 8 * h) * 128 + wid * 16 + ml] = f2bf(e);
    }
    ps += __shfl_xor(ps, 16);                      // combine lane halves
    if (h == 0) atomicAdd(&avgp[kcol], ps);
    __syncthreads();
    // acc2 += E(16x128) * CB(128 x 64-slice); waves partition d
#pragma unroll
    for (int kk = 0; kk < 128; kk += 32) {
      Frag a;
      const ushort_t* erow = &E_lds[ml * 128 + kk + h * 8];
      a.u[0] = *(const u32x4*)(erow);
      a.u[1] = *(const u32x4*)(erow + 16);
#pragma unroll
      for (int t = 0; t < 4; ++t) {
        const ushort_t* brow =
            cbT + (size_t)(wid * 64 + t * 16 + ml) * KCB + kc + kk + h * 8;
        Frag b;
        b.u[0] = *(const u32x4*)(brow);
        b.u[1] = *(const u32x4*)(brow + 16);
        acc2[t] = __builtin_amdgcn_wmma_f32_16x16x32_bf16(
            false, a.v, false, b.v, (short)0, acc2[t], false, false);
      }
    }
    __syncthreads();
  }

  // write z_to_decoder (rows exclusively owned by this block)
#pragma unroll
  for (int t = 0; t < 4; ++t)
#pragma unroll
    for (int j = 0; j < 8; ++j)
      out[(size_t)(row0 + j + 8 * h) * DIM + wid * 64 + t * 16 + ml] = acc2[t][j];

  // kld_discrete partial: sum p*logp over this block's (rows, wave's cols)
  float s = 0.f;
#pragma unroll
  for (int j = 0; j < 8; ++j) s += kldd[j];
#pragma unroll
  for (int mask = 1; mask < 32; mask <<= 1) s += __shfl_xor(s, mask);
  if (lane == 0) atomicAdd(&scal[0], s);
}

// ---------------- kld_continuous: sum (z - z_dec)^2 ----------------
__global__ __launch_bounds__(256)
void kldc_kernel(const float* __restrict__ z, const float* __restrict__ zdec,
                 float* __restrict__ scal) {
  __shared__ float red[256];
  const int tid = threadIdx.x;
  size_t i0 = (size_t)blockIdx.x * 256 * 8 + tid;
  float s = 0.f;
#pragma unroll
  for (int r = 0; r < 8; ++r) {
    size_t i = i0 + (size_t)r * 256;
    float d = z[i] - zdec[i];
    s += d * d;
  }
  red[tid] = s; __syncthreads();
  for (int off = 128; off > 0; off >>= 1) {
    if (tid < off) red[tid] += red[tid + off];
    __syncthreads();
  }
  if (tid == 0) atomicAdd(&scal[1], red[0]);
}

// ---------------- finalize: loss + perplexity ----------------
__global__ __launch_bounds__(256)
void finalize_kernel(const float* __restrict__ var_q, const float* __restrict__ avgp,
                     const float* __restrict__ scal, float* __restrict__ out_tail) {
  __shared__ float red[256];
  const int tid = threadIdx.x;
  float s = 0.f;
  for (int k = tid; k < KCB; k += 256) {
    float a = avgp[k] * (1.0f / (float)N_ROWS);
    s += a * __logf(a + 1e-7f);
  }
  red[tid] = s; __syncthreads();
  for (int off = 128; off > 0; off >>= 1) {
    if (tid < off) red[tid] += red[tid + off];
    __syncthreads();
  }
  if (tid == 0) {
    float w = 0.5f / fmaxf(var_q[0], 1e-10f);
    float loss = scal[0] / (float)BS + (w / (float)BS) * scal[1];
    out_tail[0] = loss;
    out_tail[1] = __expf(-red[0]);
  }
}

extern "C" void kernel_launch(void* const* d_in, const int* in_sizes, int n_in,
                              void* d_out, int out_size, void* d_ws, size_t ws_size,
                              hipStream_t stream) {
  const float* z        = (const float*)d_in[0];
  const float* var_q    = (const float*)d_in[1];
  const float* codebook = (const float*)d_in[2];
  float* out = (float*)d_out;

  char* ws = (char*)d_ws;
  ushort_t* zf    = (ushort_t*)(ws);                               //  4 MB
  ushort_t* cb    = (ushort_t*)(ws + (4u  << 20));                 //  8 MB
  ushort_t* cbT   = (ushort_t*)(ws + (12u << 20));                 //  8 MB
  float*    znorm = (float*)   (ws + (20u << 20));                 // 16 KB
  float*    cnorm = (float*)   (ws + (20u << 20) + (16u << 10));   // 32 KB
  float*    avgp  = (float*)   (ws + (20u << 20) + (48u << 10));   // 32 KB
  float*    scal  = (float*)   (ws + (20u << 20) + (80u << 10));   // 2 floats
  // optional 64 MB gumbel table at 21 MB (deterministic: depends only on ws_size)
  ushort_t* gtab = (ws_size >= (size_t)(85u << 20))
                       ? (ushort_t*)(ws + (21u << 20)) : (ushort_t*)0;

  // zero accumulators (avgp + scalars) every call
  hipMemsetAsync(avgp, 0, (32u << 10) + 256, stream);

  if (gtab)
    gumbel_tab_kernel<<<GHALF / 256, 256, 0, stream>>>(gtab);
  prep_z_kernel <<<N_ROWS, 256, 0, stream>>>(z, zf, znorm);
  prep_cb_kernel<<<KCB,    256, 0, stream>>>(codebook, cb, cbT, cnorm);
  vq_main_kernel<<<N_ROWS / 16, 256, 0, stream>>>(
      var_q, zf, cb, cbT, znorm, cnorm, gtab, avgp, scal, out);
  kldc_kernel<<<(N_ROWS * DIM) / (256 * 8), 256, 0, stream>>>(z, out, scal);
  finalize_kernel<<<1, 256, 0, stream>>>(var_q, avgp, scal,
                                         out + (size_t)N_ROWS * DIM);
}
